// Network_74019466379996
// MI455X (gfx1250) — matrix operand order repaired and checked by reading, not compiled, not run on hardware
//
#include <hip/hip_runtime.h>

// MI455X (gfx1250) fused SLAYER-SNN forward:
//   prep:   W1,W2 f32 -> f16 (W2 zero-padded to 32 rows)
//   layer1: WMMA GEMM (x @ W1^T) fused with psp+spike scan -> s1 f16 [n][t][h]
//   layer2: K-split WMMA GEMM (s1 @ W2^T) + LDS reduction, fused scan -> out
// Workspace use: 524288 (W1h) + 65536 (W2h) + 65,536,000 (s1) = 66,125,824 B.

#define N_BATCH 64
#define I_DIM   256
#define H_DIM   1024
#define O_DIM   18
#define T_DIM   500
#define T_CHUNKS 32  // ceil(500/16) padded to 512 t-slots

typedef __attribute__((ext_vector_type(16))) _Float16 v16h;
typedef __attribute__((ext_vector_type(8)))  float    v8f;

union F16Frag { unsigned u[8]; v16h h; };

__device__ __forceinline__ float lane_bcast(float v, int srcLane) {
  return __builtin_bit_cast(float,
      __builtin_amdgcn_ds_bpermute(srcLane << 2, __builtin_bit_cast(int, v)));
}

// Fused psp (tau=10) + spike/refractory (tau=1) scan over one 16-timestep
// C-tile. Lane layout: channel = lane%16, t = r + 8*(lane>>4) for VGPR r.
// st[4] = {pS, aS, pR, aR} (duplicated across the two half-waves).
__device__ __forceinline__ void scan16(const float cf[8], float st[4],
                                       float sp[8], int lane) {
  const float dS = 0.9048374180359595f;  // exp(-1/10)
  const float cS = 0.2718281828459045f;  // e/10
  const float dR = 0.36787944117144233f; // exp(-1)
  const float cR = 2.718281828459045f;   // e
  const float REF = -20.0f;              // -scaleRef*theta
  const float TH = 10.0f;
  const int lo = lane & 15;

  // Phase 1: lanes 0-15 advance t = 0..7 (hi lanes compute discardable values)
  float pS = st[0], aS = st[1], pR = st[2], aR = st[3];
  float s1v[8];
#pragma unroll
  for (int r = 0; r < 8; ++r) {
    aS = dS * (aS + pS);
    pS = dS * pS + cf[r];
    aR = dR * (aR + pR);
    float u = cS * aS + cR * aR - TH;
    float s = (u >= 0.0f) ? 1.0f : 0.0f;
    pR = dR * pR + REF * s;
    s1v[r] = s;
  }
  // Hand phase-1 end state (lane lo) to the upper half-wave.
  float pSx = lane_bcast(pS, lo), aSx = lane_bcast(aS, lo);
  float pRx = lane_bcast(pR, lo), aRx = lane_bcast(aR, lo);
  // Phase 2: lanes 16-31 advance t = 8..15.
  float s2v[8];
#pragma unroll
  for (int r = 0; r < 8; ++r) {
    aSx = dS * (aSx + pSx);
    pSx = dS * pSx + cf[r];
    aRx = dR * (aRx + pRx);
    float u = cS * aSx + cR * aRx - TH;
    float s = (u >= 0.0f) ? 1.0f : 0.0f;
    pRx = dR * pRx + REF * s;
    s2v[r] = s;
  }
  const bool hi = lane >= 16;
#pragma unroll
  for (int r = 0; r < 8; ++r) sp[r] = hi ? s2v[r] : s1v[r];
  // End-of-chunk state lives in lanes 16-31; rebroadcast to everyone.
  st[0] = lane_bcast(pSx, lo + 16);
  st[1] = lane_bcast(aSx, lo + 16);
  st[2] = lane_bcast(pRx, lo + 16);
  st[3] = lane_bcast(aRx, lo + 16);
}

__global__ void snn_prep_weights(const float* __restrict__ W1,
                                 const float* __restrict__ W2,
                                 _Float16* __restrict__ W1h,
                                 _Float16* __restrict__ W2h) {
  int idx = blockIdx.x * 256 + threadIdx.x;
  if (idx < H_DIM * I_DIM) W1h[idx] = (_Float16)W1[idx];
  if (idx < 32 * H_DIM) {
    int o = idx >> 10, h = idx & 1023;
    W2h[idx] = (o < O_DIM) ? (_Float16)W2[o * H_DIM + h] : (_Float16)0.0f;
  }
}

// Layer 1: grid = 64*8 blocks of 256 (8 waves). Block = (n, 128-h slab);
// wave = 16-h strip. Walks all 500 t sequentially carrying scan state.
__global__ void __launch_bounds__(256)
snn_layer1(const float* __restrict__ x, const _Float16* __restrict__ W1h,
           _Float16* __restrict__ s1) {
  __shared__ _Float16 xs[16][264];  // 16t x 256i staged tile, padded stride

  const int n = blockIdx.x >> 3;
  const int hblk = blockIdx.x & 7;
  const int wave = threadIdx.x >> 5;
  const int lane = threadIdx.x & 31;
  const int lo = lane & 15, half = lane >> 4;
  const int h0 = hblk * 128 + wave * 16;

  // Preload all B fragments (W1^T, f16, N=16 cols, K=256) — reused 500x.
  unsigned Bf[8][8];
  {
    const _Float16* wrow = W1h + (size_t)(h0 + lo) * I_DIM;
#pragma unroll
    for (int kc = 0; kc < 8; ++kc)
#pragma unroll
      for (int r = 0; r < 8; ++r)
        Bf[kc][r] = *(const unsigned*)(wrow + kc * 32 + (r & 3) * 2 + (r >> 2) * 16 + half * 8);
  }

  float st[4] = {0.f, 0.f, 0.f, 0.f};
  const int i = threadIdx.x;  // this thread stages input row i
  const size_t xbase = ((size_t)(n * I_DIM + i)) * T_DIM;

  for (int tc = 0; tc < T_CHUNKS; ++tc) {
    const int t0 = tc * 16;
    // Stage x[n, i, t0:t0+16] -> LDS f16 [t][i] (zero-pad past T).
#pragma unroll
    for (int j = 0; j < 16; ++j) {
      int t = t0 + j;
      xs[j][i] = (_Float16)((t < T_DIM) ? x[xbase + t] : 0.0f);
    }
    __syncthreads();

    v8f c = {};
#pragma unroll
    for (int kc = 0; kc < 8; ++kc) {
      F16Frag A, B;
#pragma unroll
      for (int r = 0; r < 8; ++r) {
        int K = kc * 32 + (r & 3) * 2 + (r >> 2) * 16 + half * 8;
        A.u[r] = *(const unsigned*)&xs[lo][K];
        B.u[r] = Bf[kc][r];
      }
      c = __builtin_amdgcn_wmma_f32_16x16x32_f16(
          false, A.h, false, B.h, (short)0, c, false, false);
    }
    __syncthreads();

    float cf[8];
#pragma unroll
    for (int r = 0; r < 8; ++r) cf[r] = c[r];
    float sp[8];
    scan16(cf, st, sp, lane);

    // Store spikes as f16 in [n][t][h] layout (A-friendly for layer 2).
#pragma unroll
    for (int r = 0; r < 8; ++r) {
      int t = t0 + half * 8 + r;
      if (t < T_DIM)
        s1[((size_t)(n * T_DIM + t)) * H_DIM + h0 + lo] = (_Float16)sp[r];
    }
  }
}

// Layer 2: grid = 64*2 workgroups of 8 waves. Workgroup = (n, 16-o tile);
// the K=1024 reduction is split across the 8 waves (128 h each), partial
// 16x16 f32 tiles are reduced through LDS, and wave 0 carries the sequential
// psp+spike scan state across the 32 t-chunks.
__global__ void __launch_bounds__(256)
snn_layer2(const _Float16* __restrict__ s1, const _Float16* __restrict__ W2h,
           float* __restrict__ out) {
  __shared__ float red[8][256];  // per-wave partial C tiles
  __shared__ float fin[256];     // reduced C tile

  const int n = blockIdx.x >> 1;
  const int ot = blockIdx.x & 1;
  const int w = threadIdx.x >> 5;
  const int lane = threadIdx.x & 31;
  const int lo = lane & 15, half = lane >> 4;

  // Preload this wave's B fragments (W2^T slice, K in [w*128, w*128+128)).
  unsigned Bf[4][8];
  {
    const _Float16* wrow = W2h + (size_t)(ot * 16 + lo) * H_DIM + w * 128;
#pragma unroll
    for (int kc = 0; kc < 4; ++kc)
#pragma unroll
      for (int r = 0; r < 8; ++r)
        Bf[kc][r] = *(const unsigned*)(wrow + kc * 32 + (r & 3) * 2 + (r >> 2) * 16 + half * 8);
  }

  float st[4] = {0.f, 0.f, 0.f, 0.f};

  for (int tc = 0; tc < T_CHUNKS; ++tc) {
    const int tA = tc * 16 + lo;  // A-matrix row (M = lane%16)
    const bool tok = (tA < T_DIM);
    const _Float16* arow =
        s1 + ((size_t)(n * T_DIM + (tok ? tA : 0))) * H_DIM + w * 128;

    v8f c = {};
#pragma unroll
    for (int kc = 0; kc < 4; ++kc) {
      F16Frag A, B;
#pragma unroll
      for (int r = 0; r < 8; ++r) {
        int K = kc * 32 + (r & 3) * 2 + (r >> 2) * 16 + half * 8;
        unsigned av = *(const unsigned*)(arow + K);
        A.u[r] = tok ? av : 0u;
        B.u[r] = Bf[kc][r];
      }
      c = __builtin_amdgcn_wmma_f32_16x16x32_f16(
          false, A.h, false, B.h, (short)0, c, false, false);
    }

    // Publish partial tile, then reduce across the 8 waves.
#pragma unroll
    for (int r = 0; r < 8; ++r) red[w][r * 32 + lane] = c[r];
    __syncthreads();
    {
      const int e = threadIdx.x;  // one C element per thread
      float a = 0.0f;
#pragma unroll
      for (int ww = 0; ww < 8; ++ww) a += red[ww][e];
      fin[e] = a;
    }
    __syncthreads();

    if (w == 0) {  // full wave: EXEC all-ones inside, bpermute is safe
      float cf[8];
#pragma unroll
      for (int r = 0; r < 8; ++r) cf[r] = fin[r * 32 + lane];
      float sp[8];
      scan16(cf, st, sp, lane);

      const int o = ot * 16 + lo;
      if (o < O_DIM) {
#pragma unroll
        for (int r = 0; r < 8; ++r) {
          int t = tc * 16 + half * 8 + r;
          if (t < T_DIM) out[((size_t)n * O_DIM + o) * T_DIM + t] = sp[r];
        }
      }
    }
  }
}

extern "C" void kernel_launch(void* const* d_in, const int* in_sizes, int n_in,
                              void* d_out, int out_size, void* d_ws,
                              size_t ws_size, hipStream_t stream) {
  const float* x = (const float*)d_in[0];   // [64,256,500]
  const float* W1 = (const float*)d_in[1];  // [1024,256]
  const float* W2 = (const float*)d_in[2];  // [18,1024]

  char* ws = (char*)d_ws;
  _Float16* W1h = (_Float16*)ws;                       // 524288 B
  _Float16* W2h = (_Float16*)(ws + 524288);            // 65536 B
  _Float16* s1 = (_Float16*)(ws + 524288 + 65536);     // 65,536,000 B

  snn_prep_weights<<<1024, 256, 0, stream>>>(W1, W2, W1h, W2h);
  snn_layer1<<<N_BATCH * 8, 256, 0, stream>>>(x, W1h, s1);
  snn_layer2<<<N_BATCH * 2, 256, 0, stream>>>(s1, W2h, (float*)d_out);
}
